// UltimateTNN_901943132739
// MI455X (gfx1250) — matrix operand order repaired and checked by reading, not compiled, not run on hardware
//
#include <hip/hip_runtime.h>
#include <math.h>

// Model dims
#define H    1024
#define HF   512
#define VSZ  32000
#define SS   512
#define LL   4
#define NBK  8
#define MTOT 4096   // B*S

typedef __attribute__((ext_vector_type(16))) __bf16 v16bf;
typedef __attribute__((ext_vector_type(8)))  float  v8f;

__device__ __forceinline__ unsigned short f32_bf16(float f) {
    __bf16 h = (__bf16)f;                       // RNE fptrunc
    return __builtin_bit_cast(unsigned short, h);
}

union FragU { uint4 q[2]; v16bf v; };

// CDNA5 async copy: 16B per lane, global -> LDS, tracked by ASYNCcnt
__device__ __forceinline__ void async_ld16(unsigned lds_byte_addr,
                                           unsigned long long gaddr) {
    asm volatile("global_load_async_to_lds_b128 %0, %1, off"
                 :: "v"(lds_byte_addr), "v"(gaddr) : "memory");
}
__device__ __forceinline__ void wait_async0() {
    asm volatile("s_wait_asynccnt 0x0" ::: "memory");
}

#define APAD  40           // bf16 elems per LDS row: 80B rows (16B aligned, conflict-free)
#define TILEE (128 * APAD) // ushorts per tile buffer

// ---------------------------------------------------------------------------
// bf16 WMMA GEMM, both operands bf16, B pre-transposed: Bt[N,Kd].
// D = epilogue(A[M,Kd] x Bt^T + bias[idx])   (MODE is compile-time)
// MODE 0: outf = res + 0.5*(acc+bias)*(1+sigmoid(tg[idx])*tf)   (attn)
// MODE 1: outh = bf16(gelu_exact(acc+bias))                      (ff1)
// MODE 2: outf = (res + 0.5*(acc+bias)) * fscale                 (ff2, +1.5x)
// MODE 3: outf = acc + bias                                      (lm head)
// ---------------------------------------------------------------------------
template <int MODE>
__global__ __launch_bounds__(256)
void gemm_bf16(const unsigned short* __restrict__ A,
               const unsigned short* __restrict__ Bt,
               const float* __restrict__ biasbase, int bstride,
               const int* __restrict__ idxp, int slot,
               int N, int Kd,
               const float* __restrict__ res,
               float* __restrict__ outf,
               unsigned short* __restrict__ outh,
               const float* __restrict__ tgbase,
               const float* __restrict__ tf,
               float fscale)
{
    __shared__ unsigned short As[2 * TILEE];  // double-buffered A tile 128x32
    __shared__ unsigned short Bs[2 * TILEE];  // double-buffered B tile 128x32 ([n][k])

    const int tid   = threadIdx.x;
    const int lane  = tid & 31;
    const int w     = tid >> 5;
    const int waveM = w >> 2;       // 0..1 -> 64 rows
    const int waveN = w & 3;        // 0..3 -> 32 cols
    const int mtile = blockIdx.y;
    const int ntile = blockIdx.x;

    const int idx = idxp ? idxp[slot] : 0;
    const float* bias = biasbase + (long)idx * bstride;

    // staging: each thread owns one row's 16-elem half (32B) per tile
    const int ar = tid >> 1;            // 0..127
    const int ah = (tid & 1) * 16;      // element offset 0 / 16
    const unsigned short* Arow = A  + (long)(mtile * 128 + ar) * Kd + ah;
    const unsigned short* Brow = Bt + (long)(ntile * 128 + ar) * Kd + ah;
    const unsigned aLds = (unsigned)(size_t)&As[ar * APAD + ah];
    const unsigned bLds = (unsigned)(size_t)&Bs[ar * APAD + ah];

    v8f acc[4][2];
    for (int i = 0; i < 4; ++i)
        for (int j = 0; j < 2; ++j)
            for (int r = 0; r < 8; ++r) acc[i][j][r] = 0.0f;

    const int nK = Kd >> 5;
    const unsigned bufB = TILEE * 2;    // bytes per buffer

    // prologue: stage tile 0 into buffer 0
    async_ld16(aLds,      (unsigned long long)(size_t)(Arow));
    async_ld16(aLds + 16, (unsigned long long)(size_t)(Arow + 8));
    async_ld16(bLds,      (unsigned long long)(size_t)(Brow));
    async_ld16(bLds + 16, (unsigned long long)(size_t)(Brow + 8));
    wait_async0();
    __syncthreads();

    const int khalf = lane >> 4;
    const int rl    = lane & 15;

    for (int kc = 0; kc < nK; ++kc) {
        const int p = kc & 1;
        const bool more = (kc + 1) < nK;
        if (more) {  // async-stage next tile into the other buffer
            const unsigned o = (p ^ 1) * bufB;
            const unsigned short* an = Arow + (kc + 1) * 32;
            const unsigned short* bn = Brow + (kc + 1) * 32;
            async_ld16(aLds + o,      (unsigned long long)(size_t)(an));
            async_ld16(aLds + o + 16, (unsigned long long)(size_t)(an + 8));
            async_ld16(bLds + o,      (unsigned long long)(size_t)(bn));
            async_ld16(bLds + o + 16, (unsigned long long)(size_t)(bn + 8));
        }
        // fragment gathers (ISA 16-bit layouts):
        // A lane<16: K 0..7,16..23 ; lane>=16: K 8..15,24..31
        // B lane<16: K 0..7,16..23 ; lane>=16: K 8..15,24..31
        const unsigned short* Ab = &As[p * TILEE];
        const unsigned short* Bb = &Bs[p * TILEE];
        FragU a[4], b[2];
        #pragma unroll
        for (int ms = 0; ms < 4; ++ms) {
            const int row = waveM * 64 + ms * 16 + rl;
            a[ms].q[0] = *(const uint4*)&Ab[row * APAD + khalf * 8];
            a[ms].q[1] = *(const uint4*)&Ab[row * APAD + 16 + khalf * 8];
        }
        #pragma unroll
        for (int ns = 0; ns < 2; ++ns) {
            const int col = waveN * 32 + ns * 16 + rl;
            b[ns].q[0] = *(const uint4*)&Bb[col * APAD + khalf * 8];
            b[ns].q[1] = *(const uint4*)&Bb[col * APAD + 16 + khalf * 8];
        }
        #pragma unroll
        for (int ms = 0; ms < 4; ++ms)
            #pragma unroll
            for (int ns = 0; ns < 2; ++ns)
                acc[ms][ns] = __builtin_amdgcn_wmma_f32_16x16x32_bf16(
                    false, a[ms].v, false, b[ns].v, (short)0, acc[ms][ns],
                    false, false);
        if (more) {
            wait_async0();       // next tile resident
            __syncthreads();     // visible; previous buffer's readers done
        }
    }

    // Epilogue. C/D layout: VGPR r -> M = r + 8*(lane>=16), N = lane%16
    const int rh = (lane >> 4) * 8;
    // per-lane column constants (2 columns per lane)
    float bv[2], gv[2];
    #pragma unroll
    for (int ns = 0; ns < 2; ++ns) {
        const int col = ntile * 128 + waveN * 32 + ns * 16 + rl;
        bv[ns] = bias[col];
        if (MODE == 0) {
            const float t = tgbase[(long)idx * H + col];
            gv[ns] = 1.0f + (1.0f / (1.0f + __expf(-t))) * tf[col];
        }
    }
    #pragma unroll
    for (int ms = 0; ms < 4; ++ms) {
        #pragma unroll
        for (int ns = 0; ns < 2; ++ns) {
            const int col = ntile * 128 + waveN * 32 + ns * 16 + rl;
            #pragma unroll
            for (int r = 0; r < 8; ++r) {
                const int row = mtile * 128 + waveM * 64 + ms * 16 + rh + r;
                const long o  = (long)row * N + col;
                const float v = acc[ms][ns][r] + bv[ns];
                if (MODE == 0)
                    outf[o] = res[o] + 0.5f * v * gv[ns];
                else if (MODE == 1)
                    outh[o] = f32_bf16(0.5f * v * (1.0f + erff(v * 0.70710678f)));
                else if (MODE == 2)
                    outf[o] = (res[o] + 0.5f * v) * fscale;
                else
                    outf[o] = v;
            }
        }
    }
}

// ---------------------------------------------------------------------------
// Transpose + f32->bf16 weight conversion: src[K,N] (+expert idx) -> dst[N,K].
// ---------------------------------------------------------------------------
__global__ __launch_bounds__(256)
void tcvt_kernel(const float* __restrict__ srcbase, long sstride,
                 const int* __restrict__ idxp, int slot,
                 unsigned short* __restrict__ dst, int K, int N)
{
    __shared__ float t[32][33];
    const int idx = idxp ? idxp[slot] : 0;
    const float* src = srcbase + (long)idx * sstride;
    const int tx = threadIdx.x & 31, ty = threadIdx.x >> 5;  // 32 x 8
    const int n0 = blockIdx.x * 32, k0 = blockIdx.y * 32;
    #pragma unroll
    for (int j = 0; j < 4; ++j)
        t[ty + 8 * j][tx] = src[(long)(k0 + ty + 8 * j) * N + n0 + tx];
    __syncthreads();
    #pragma unroll
    for (int j = 0; j < 4; ++j)
        dst[(long)(n0 + ty + 8 * j) * K + k0 + tx] = f32_bf16(t[tx][ty + 8 * j]);
}

// ---------------------------------------------------------------------------
// Fused LayerNorm -> bf16 (plain convert when gbase==nullptr).
// ---------------------------------------------------------------------------
__global__ __launch_bounds__(256)
void ln_kernel(const float* __restrict__ X,
               const float* __restrict__ gbase, const float* __restrict__ bbase,
               const int* __restrict__ idxp, int slot,
               unsigned short* __restrict__ out)
{
    __shared__ float s1[256], s2[256];
    const int row = blockIdx.x;
    const int tid = threadIdx.x;
    const float* xr = X + (long)row * H;
    float a = 0.f, b = 0.f;
    #pragma unroll
    for (int j = 0; j < H / 256; ++j) {
        float v = xr[tid + j * 256];
        a += v; b += v * v;
    }
    s1[tid] = a; s2[tid] = b;
    __syncthreads();
    for (int off = 128; off > 0; off >>= 1) {
        if (tid < off) { s1[tid] += s1[tid + off]; s2[tid] += s2[tid + off]; }
        __syncthreads();
    }
    const float m    = s1[0] * (1.0f / H);
    const float var  = s2[0] * (1.0f / H) - m * m;
    const float rstd = rsqrtf(var + 1e-5f);
    unsigned short* orow = out + (long)row * H;
    if (gbase) {
        const int idx = idxp ? idxp[slot] : 0;
        const float* g  = gbase + (long)idx * H;
        const float* be = bbase + (long)idx * H;
        for (int j = 0; j < H / 256; ++j) {
            const int c = tid + j * 256;
            orow[c] = f32_bf16((xr[c] - m) * rstd * g[c] + be[c]);
        }
    } else {
        for (int j = 0; j < H / 256; ++j) {
            const int c = tid + j * 256;
            orow[c] = f32_bf16(xr[c]);
        }
    }
}

// Embedding + positional add
__global__ __launch_bounds__(256)
void embed_kernel(const int* __restrict__ ids, const float* __restrict__ emb,
                  const float* __restrict__ pos, float* __restrict__ X)
{
    const long i = (long)blockIdx.x * 256 + threadIdx.x;
    const int h  = (int)(i & (H - 1));
    const int bs = (int)(i >> 10);
    const int sr = bs & (SS - 1);
    const int id = ids[bs];
    X[i] = emb[(long)id * H + h] + pos[(long)sr * H + h];
}

// Mean-pool batch row 0 over sequence
__global__ __launch_bounds__(256)
void pool0_kernel(const float* __restrict__ X, float* __restrict__ pooled)
{
    const int h = blockIdx.x * 256 + threadIdx.x;
    float s = 0.f;
    for (int i = 0; i < SS; ++i) s += X[(long)i * H + h];
    pooled[h] = s * (1.0f / SS);
}

// Router: 8 waves -> 8 logits; thread 0 writes top-2 expert indices
__global__ __launch_bounds__(256)
void router_kernel(const float* __restrict__ pooled,
                   const float* __restrict__ selw, const float* __restrict__ selb,
                   int* __restrict__ topidx)
{
    __shared__ float adj[NBK];
    const int tid = threadIdx.x, lane = tid & 31, w = tid >> 5;
    float s = 0.f;
    for (int j = lane; j < H; j += 32) s += pooled[j] * selw[j * NBK + w];
    for (int off = 16; off > 0; off >>= 1) s += __shfl_down(s, off, 32);
    if (lane == 0) {
        const float sc = 1.0f / (1.0f + __expf(-(s + selb[w])));
        adj[w] = sc * 0.7f + 0.15f;   // + PRIORITY(0.5)*0.3
    }
    __syncthreads();
    if (tid == 0) {
        int i0 = 0; float b0 = adj[0];
        for (int i = 1; i < NBK; ++i) if (adj[i] > b0) { b0 = adj[i]; i0 = i; }
        int i1 = -1; float b1 = -1e30f;
        for (int i = 0; i < NBK; ++i)
            if (i != i0 && adj[i] > b1) { b1 = adj[i]; i1 = i; }
        topidx[0] = i0; topidx[1] = i1;
    }
}

// ---------------------------------------------------------------------------
extern "C" void kernel_launch(void* const* d_in, const int* in_sizes, int n_in,
                              void* d_out, int out_size, void* d_ws, size_t ws_size,
                              hipStream_t stream)
{
    const int*   ids    = (const int*)  d_in[0];
    const float* emb    = (const float*)d_in[1];
    const float* pos    = (const float*)d_in[2];
    const float* tf     = (const float*)d_in[3];
    const float* lm_w   = (const float*)d_in[4];
    const float* lm_b   = (const float*)d_in[5];
    const float* sel_w  = (const float*)d_in[6];
    const float* sel_b  = (const float*)d_in[7];
    const float* g1     = (const float*)d_in[8];
    const float* b1     = (const float*)d_in[9];
    const float* attn_w = (const float*)d_in[10];
    const float* attn_b = (const float*)d_in[11];
    const float* g2     = (const float*)d_in[12];
    const float* b2     = (const float*)d_in[13];
    const float* ff1_w  = (const float*)d_in[14];
    const float* ff1_b  = (const float*)d_in[15];
    const float* ff2_w  = (const float*)d_in[16];
    const float* ff2_b  = (const float*)d_in[17];
    const float* tg     = (const float*)d_in[18];
    float* out = (float*)d_out;

    const size_t MB = 1024 * 1024;
    char* ws = (char*)d_ws;
    float*          x      = (float*)         (ws);              // 16 MB
    float*          hm     = (float*)         (ws + 16 * MB);    // 16 MB
    unsigned short* xn     = (unsigned short*)(ws + 32 * MB);    //  8 MB
    unsigned short* gbuf   = (unsigned short*)(ws + 40 * MB);    //  4 MB
    float*          pooled = (float*)         (ws + 44 * MB);    //  4 KB
    int*            topidx = (int*)           (ws + 44 * MB + 4096);
    unsigned short* lm_wt  = (unsigned short*)(ws + 45 * MB);    // 64 MB [V,H]
    unsigned short* wat[2] = { (unsigned short*)(ws + 109 * MB),
                               (unsigned short*)(ws + 111 * MB) };  // [H,H]
    unsigned short* wf1[2] = { (unsigned short*)(ws + 113 * MB),
                               (unsigned short*)(ws + 114 * MB) };  // [HF,H]
    unsigned short* wf2[2] = { (unsigned short*)(ws + 115 * MB),
                               (unsigned short*)(ws + 116 * MB) };  // [H,HF]

    // LM head weights: transpose+convert once
    tcvt_kernel<<<dim3(VSZ / 32, H / 32), 256, 0, stream>>>(
        lm_w, 0, nullptr, 0, lm_wt, H, VSZ);

    embed_kernel<<<(MTOT * H) / 256, 256, 0, stream>>>(ids, emb, pos, x);

    for (int l = 0; l < LL; ++l) {
        pool0_kernel<<<H / 256, 256, 0, stream>>>(x, pooled);
        router_kernel<<<1, 256, 0, stream>>>(pooled,
                                             sel_w + (long)l * H * NBK,
                                             sel_b + (long)l * NBK, topidx);
        for (int k = 0; k < 2; ++k) {
            // convert+transpose the selected expert's weights (device-side idx)
            tcvt_kernel<<<dim3(H / 32, H / 32), 256, 0, stream>>>(
                attn_w + (long)l * NBK * H * H, (long)H * H, topidx, k, wat[k], H, H);
            tcvt_kernel<<<dim3(HF / 32, H / 32), 256, 0, stream>>>(
                ff1_w + (long)l * NBK * H * HF, (long)H * HF, topidx, k, wf1[k], H, HF);
            tcvt_kernel<<<dim3(H / 32, HF / 32), 256, 0, stream>>>(
                ff2_w + (long)l * NBK * HF * H, (long)HF * H, topidx, k, wf2[k], HF, H);

            // hm = x + 0.5*gate*(LN1(x) @ attn_w + attn_b)
            ln_kernel<<<MTOT, 256, 0, stream>>>(x, g1 + (long)l * NBK * H,
                                                b1 + (long)l * NBK * H, topidx, k, xn);
            gemm_bf16<0><<<dim3(H / 128, MTOT / 128), 256, 0, stream>>>(
                xn, wat[k], attn_b + (long)l * NBK * H, H, topidx, k,
                H, H, x, hm, nullptr, tg + (long)l * NBK * H, tf, 1.0f);
            // g = gelu(LN2(hm) @ ff1_w + ff1_b)
            ln_kernel<<<MTOT, 256, 0, stream>>>(hm, g2 + (long)l * NBK * H,
                                                b2 + (long)l * NBK * H, topidx, k, xn);
            gemm_bf16<1><<<dim3(HF / 128, MTOT / 128), 256, 0, stream>>>(
                xn, wf1[k], ff1_b + (long)l * NBK * HF, HF, topidx, k,
                HF, H, nullptr, nullptr, gbuf, nullptr, nullptr, 1.0f);
            // x = (hm + 0.5*(g @ ff2_w + ff2_b)) * (k==1 ? 1.5 : 1.0)
            gemm_bf16<2><<<dim3(H / 128, MTOT / 128), 256, 0, stream>>>(
                gbuf, wf2[k], ff2_b + (long)l * NBK * H, H, topidx, k,
                H, HF, hm, x, nullptr, nullptr, nullptr,
                (k == 1) ? 1.5f : 1.0f);
        }
    }

    // LM head: out = x @ lm_w + lm_b
    ln_kernel<<<MTOT, 256, 0, stream>>>(x, nullptr, nullptr, nullptr, 0, xn);
    gemm_bf16<3><<<dim3(VSZ / 128, MTOT / 128), 256, 0, stream>>>(
        xn, lm_wt, lm_b, 0, nullptr, 0, VSZ, H,
        nullptr, out, nullptr, nullptr, nullptr, 1.0f);
}
